// CausalAttention_6193342841181
// MI455X (gfx1250) — compile-verified, matrix operand order checked by reading, not run on hardware
//
#include <hip/hip_runtime.h>
#include <hip/hip_bf16.h>

// ---------------------------------------------------------------------------
// CDNA5 / gfx1250 fused causal attention (ALiBi).
// Compute-bound (~412 GFLOP vs ~350 MB of HBM traffic): everything runs
// through v_wmma_f32_16x16x32_bf16 (wave32 WMMA, f32 accumulate).
// Flash-style streaming softmax avoids the 1.07 GB S x S score tensor.
// K tiles are staged into LDS by the Tensor Data Mover (tensor_load_to_lds,
// TENSORcnt) with D# padding matching the WMMA fragment row stride.
// ---------------------------------------------------------------------------

using bf16 = __bf16;
typedef __attribute__((ext_vector_type(16))) __bf16 v16bf;
typedef __attribute__((ext_vector_type(8)))  __bf16 v8bf;
typedef __attribute__((ext_vector_type(2)))  __bf16 v2bf;
typedef __attribute__((ext_vector_type(8)))  float  v8f;
typedef __attribute__((ext_vector_type(4)))  unsigned int u32x4;
typedef __attribute__((ext_vector_type(8)))  int i32x8;
typedef __attribute__((ext_vector_type(4)))  int i32x4;

constexpr int Bc  = 4;
constexpr int Sc  = 2048;
constexpr int Dc  = 2048;
constexpr int NHc = 16;
constexpr int HDc = 128;          // head dim

// --- TDM availability (device pass only; host pass falls back cleanly) ----
#if defined(__has_builtin)
#if __has_builtin(__builtin_amdgcn_tensor_load_to_lds) && \
    __has_builtin(__builtin_amdgcn_s_wait_tensorcnt)
#define HAVE_TDM 1
#endif
#endif
#ifndef HAVE_TDM
#define HAVE_TDM 0
#endif

// -------------------------------------------------------------------------
// A/B fragment loader for V_WMMA_F32_16X16X32_BF16.
// ISA 16-bit A-matrix 16x32 layout (CDNA5 ISA):
//   lanes 0-15  (half=0): row M = lane,    K = {0..7} U {16..23}
//   lanes 16-31 (half=1): row M = lane-16, K = {8..15} U {24..31}
// With row-major [row][k] storage (rows padded to stay 16B-aligned) this is
// exactly two contiguous ds_load_b128 per lane.  B fragments use the same
// loader on a transposed [n][k] tile (B^T), mirroring the layout.
// -------------------------------------------------------------------------
static __device__ __forceinline__ v16bf load_frag(const bf16* __restrict__ row, int half) {
  const v8bf lo = *reinterpret_cast<const v8bf*>(row + (half << 3));
  const v8bf hi = *reinterpret_cast<const v8bf*>(row + 16 + (half << 3));
  return __builtin_shufflevector(lo, hi, 0, 1, 2, 3, 4, 5, 6, 7,
                                         8, 9, 10, 11, 12, 13, 14, 15);
}

static __device__ __forceinline__ v8f wmma_bf16(v16bf a, v16bf b, v8f c) {
  return __builtin_amdgcn_wmma_f32_16x16x32_bf16(
      /*neg_a=*/false, a, /*neg_b=*/false, b,
      /*c_mod=*/(short)0, c, /*reuse_a=*/false, /*reuse_b=*/false);
}

#if HAVE_TDM
// ---------------------------------------------------------------------------
// Tensor Data Mover: 2D tile load Global -> LDS.
//   tile  : 64 rows x 128 bf16 cols, global row stride 2048 elements
//   LDS   : padded to 136-element rows via D# pad (interval 64 DW, amount 4 DW)
// D# layout per CDNA5 ISA.
// ---------------------------------------------------------------------------
static __device__ __forceinline__ void tdm_load_k_tile(unsigned lds_addr,
                                                       const void* gaddr) {
  const unsigned long long ga = (unsigned long long)(uintptr_t)gaddr;
  const u32x4 g0 = {
      1u,                                               // count=1, user mode
      lds_addr,                                         // lds_addr
      (unsigned)(ga & 0xFFFFFFFFu),                     // global_addr[31:0]
      (unsigned)((ga >> 32) & 0x01FFFFFFu) | (2u << 30) // addr[56:32] | type=2
  };
  const i32x8 g1 = {
      (int)((1u << 16) |        // data_size = 1 -> 2 bytes (bf16)
            (1u << 20) |        // pad_enable
            (5u << 22) |        // pad_interval: every 64 DWORDs (256B row)
            (3u << 25)),        // pad_amount : 4 DWORDs (16B -> 136-elem rows)
      (int)(128u << 16),        // tensor_dim0 = 128 (low16 at bits 63:48)
      (int)(64u << 16),         // dim0 hi=0 | tensor_dim1 = 64 (bits 95:80)
      (int)(128u << 16),        // dim1 hi=0 | tile_dim0 = 128 (bits 127:112)
      64,                       // tile_dim1 = 64, tile_dim2 = 0
      2048,                     // tensor_dim0_stride = 2048 (low 32)
      0, 0                      // stride hi / tensor_dim1_stride (unused, 2D)
  };
  const i32x4 gz = {0, 0, 0, 0};
#if __clang_major__ >= 23
  const i32x8 gz8 = {0, 0, 0, 0, 0, 0, 0, 0};
  __builtin_amdgcn_tensor_load_to_lds(g0, g1, gz, gz, gz8, 0);
#else
  __builtin_amdgcn_tensor_load_to_lds(g0, g1, gz, gz, 0);
#endif
}
#endif  // HAVE_TDM

// ---------------------------------------------------------------------------
// GEMM: C[M,N] = A[M,K] * B[K,N].  A fp32 or bf16, B fp32, C fp32 or bf16.
// 128x128 block tile, BK=32 (one WMMA K-step), 8 waves of 64x32 each.
// Staging converts to bf16 in packed pairs (one ds_store_b32 per 2 elems).
// ---------------------------------------------------------------------------
template <bool A_IS_BF16, bool OUT_IS_BF16>
__global__ __launch_bounds__(256) void wmma_gemm_kernel(
    const void* __restrict__ Ap, const float* __restrict__ Bp,
    void* __restrict__ Cp, int M, int N, int K) {
  (void)M;
  constexpr int LDT = 48;  // 32 + 16 pad: 96B row stride keeps 16B alignment

  __shared__ alignas(16) bf16 As[128 * LDT];  // [m][k]
  __shared__ alignas(16) bf16 Bs[128 * LDT];  // [n][k]  (B transposed)

  const int tid  = threadIdx.x;
  const int wave = tid >> 5;
  const int lane = tid & 31;
  const int half = lane >> 4;
  const int r    = lane & 15;
  const int wm   = (wave & 1) * 64;   // wave row offset in block tile
  const int wn   = (wave >> 1) * 32;  // wave col offset in block tile

  const int m0g = blockIdx.y * 128;
  const int n0g = blockIdx.x * 128;

  v8f acc[4][2];
#pragma unroll
  for (int im = 0; im < 4; ++im)
#pragma unroll
    for (int in = 0; in < 2; ++in) acc[im][in] = v8f{0.f, 0.f, 0.f, 0.f, 0.f, 0.f, 0.f, 0.f};

  const float* Af32 = reinterpret_cast<const float*>(Ap);
  const bf16*  Abf  = reinterpret_cast<const bf16*>(Ap);

  for (int kt = 0; kt < K; kt += 32) {
    if (kt + 32 < K)  // hint next B slab into caches (global_prefetch_b8)
      __builtin_prefetch(Bp + (size_t)(kt + 32) * N + n0g + (tid & 127), 0, 1);

    // --- stage A tile (128x32) as bf16 pairs: 2048 pairs / 256 threads ---
#pragma unroll
    for (int i = 0; i < 8; ++i) {
      const int idx = (i << 8) + tid;
      const int rr = idx >> 4, cp = (idx & 15) << 1;
      const size_t g = (size_t)(m0g + rr) * K + kt + cp;
      if constexpr (A_IS_BF16) {
        *reinterpret_cast<unsigned*>(&As[rr * LDT + cp]) =
            *reinterpret_cast<const unsigned*>(Abf + g);
      } else {
        const float2 fv = *reinterpret_cast<const float2*>(Af32 + g);
        const v2bf pk = {(bf16)fv.x, (bf16)fv.y};
        *reinterpret_cast<v2bf*>(&As[rr * LDT + cp]) = pk;
      }
    }
    // --- stage B tile (32x128) transposed -> Bs[n][k], packed k-pairs ---
#pragma unroll
    for (int i = 0; i < 8; ++i) {
      const int idx = (i << 8) + tid;
      const int k2 = (idx >> 7) << 1, nn = idx & 127;
      const float b0 = Bp[(size_t)(kt + k2) * N + n0g + nn];
      const float b1 = Bp[(size_t)(kt + k2 + 1) * N + n0g + nn];
      const v2bf pk = {(bf16)b0, (bf16)b1};
      *reinterpret_cast<v2bf*>(&Bs[nn * LDT + k2]) = pk;
    }
    __syncthreads();

    v16bf a[4], b[2];
#pragma unroll
    for (int im = 0; im < 4; ++im)
      a[im] = load_frag(&As[(wm + im * 16 + r) * LDT], half);
#pragma unroll
    for (int in = 0; in < 2; ++in)
      b[in] = load_frag(&Bs[(wn + in * 16 + r) * LDT], half);
#pragma unroll
    for (int im = 0; im < 4; ++im)
#pragma unroll
      for (int in = 0; in < 2; ++in) acc[im][in] = wmma_bf16(a[im], b[in], acc[im][in]);
    __syncthreads();
  }

  // C layout: lane r = column, VGPR g = row (g + 8*half)
#pragma unroll
  for (int im = 0; im < 4; ++im)
#pragma unroll
    for (int in = 0; in < 2; ++in)
#pragma unroll
      for (int g = 0; g < 8; ++g) {
        const int m = m0g + wm + im * 16 + g + (half << 3);
        const int n = n0g + wn + in * 16 + r;
        const size_t o = (size_t)m * N + n;
        if (OUT_IS_BF16) reinterpret_cast<bf16*>(Cp)[o] = (bf16)acc[im][in][g];
        else             reinterpret_cast<float*>(Cp)[o] = acc[im][in][g];
      }
}

// ---------------------------------------------------------------------------
// Flash attention with causal mask + ALiBi per-key bias.
// Grid: (B * NH * S/64) workgroups of 128 threads (4 waves).
// Each wave owns a 16-query strip; Q fragments stay in registers; K tiles are
// DMA'd into LDS by the TDM; V^T tiles staged manually (TDM can't transpose);
// P round-trips through LDS into A-fragment layout.
// ---------------------------------------------------------------------------
__global__ __launch_bounds__(128) void flash_attn_kernel(
    const bf16* __restrict__ Qg, const bf16* __restrict__ Kg,
    const bf16* __restrict__ Vg, bf16* __restrict__ Og) {
  constexpr int LDK = 136;  // 128 + 8 pad (272B rows, 16B aligned) == TDM pad
  constexpr int LDV = 72;   // 64 + 8 pad  (144B rows, 16B aligned)

  __shared__ alignas(16) bf16 Kt[64 * LDK];   // [key][d]   == B^T for Q*K^T
  __shared__ alignas(16) bf16 VT[128 * LDV];  // [d][key]   == B^T for P*V
  __shared__ alignas(16) bf16 Pb[64 * LDV];   // [q][key]   P in A layout

  const int tid  = threadIdx.x;
  const int wave = tid >> 5;
  const int lane = tid & 31;
  const int half = lane >> 4;
  const int r    = lane & 15;

  const int qt = blockIdx.x & 31;          // S/64 = 32 query tiles
  const int h  = (blockIdx.x >> 5) & 15;   // head
  const int bb = blockIdx.x >> 9;          // batch

  const float kScale = 0.08838834764831845f;           // 1/sqrt(128)
  const float slope  = exp2f(-0.5f * (float)(h + 1));  // ALiBi, NH=16 pow2

  // Q fragments for this wave's 16 rows: 4 K-steps covering HD=128
  v16bf aQ[4];
  {
    const size_t qrow = (size_t)(bb * Sc + qt * 64 + wave * 16 + r) * Dc + h * HDc;
#pragma unroll
    for (int kidx = 0; kidx < 4; ++kidx) aQ[kidx] = load_frag(Qg + qrow + kidx * 32, half);
  }

  v8f   Oacc[8];
  float mrow[8], lrow[8];
#pragma unroll
  for (int t = 0; t < 8; ++t) Oacc[t] = v8f{0.f, 0.f, 0.f, 0.f, 0.f, 0.f, 0.f, 0.f};
#pragma unroll
  for (int g = 0; g < 8; ++g) { mrow[g] = -3.0e38f; lrow[g] = 0.f; }

  for (int kb = 0; kb <= qt; ++kb) {
    const int  kk   = kb * 64;
    const bool diag = (kb == qt);

    // --- stage K tile [64 keys][128 d] row-major into Kt via TDM ---
#if HAVE_TDM
    if (wave == 0) {  // one wave issues the DMA; EXEC ignored by TDM
      tdm_load_k_tile((unsigned)(uintptr_t)(&Kt[0]),
                      Kg + (size_t)(bb * Sc + kk) * Dc + h * HDc);
      __builtin_amdgcn_s_wait_tensorcnt(0);
    }
#else
#pragma unroll
    for (int i = 0; i < 16; ++i) {
      const int idx = i * 128 + tid;
      const int row = idx >> 5, grp = idx & 31;
      *reinterpret_cast<unsigned long long*>(&Kt[row * LDK + grp * 4]) =
          *reinterpret_cast<const unsigned long long*>(
              Kg + (size_t)(bb * Sc + kk + row) * Dc + h * HDc + grp * 4);
    }
#endif
    // --- stage V tile transposed: VT[d][key] ---
#pragma unroll
    for (int i = 0; i < 16; ++i) {
      const int idx = i * 128 + tid;
      const int key = idx >> 5, dg = idx & 31;
      bf16 tmp[4];
      *reinterpret_cast<unsigned long long*>(tmp) =
          *reinterpret_cast<const unsigned long long*>(
              Vg + (size_t)(bb * Sc + kk + key) * Dc + h * HDc + dg * 4);
#pragma unroll
      for (int e = 0; e < 4; ++e) VT[(dg * 4 + e) * LDV + key] = tmp[e];
    }
    __syncthreads();

    // --- S = Q * K^T : 4 column tiles x 4 K-steps = 16 WMMAs ---
    v8f sacc[4];
#pragma unroll
    for (int j = 0; j < 4; ++j) sacc[j] = v8f{0.f, 0.f, 0.f, 0.f, 0.f, 0.f, 0.f, 0.f};
#pragma unroll
    for (int j = 0; j < 4; ++j)
#pragma unroll
      for (int kidx = 0; kidx < 4; ++kidx)
        sacc[j] = wmma_bf16(aQ[kidx],
                            load_frag(&Kt[(j * 16 + r) * LDK + kidx * 32], half),
                            sacc[j]);

    // --- scale + ALiBi + causal mask + online softmax (per row g,half) ---
#pragma unroll
    for (int g = 0; g < 8; ++g) {
      const int qglob = qt * 64 + wave * 16 + g + (half << 3);
      float mx = -3.0e38f;
#pragma unroll
      for (int j = 0; j < 4; ++j) {
        const int key = kk + j * 16 + r;
        float s = sacc[j][g] * kScale - slope * (float)(Sc - 1 - key);
        if (diag && key > qglob) s = -3.0e38f;
        sacc[j][g] = s;
        mx = fmaxf(mx, s);
      }
      // reduce over the 16 lanes holding this row (xor<16 stays in half-group)
      mx = fmaxf(mx, __shfl_xor(mx, 1, 32));
      mx = fmaxf(mx, __shfl_xor(mx, 2, 32));
      mx = fmaxf(mx, __shfl_xor(mx, 4, 32));
      mx = fmaxf(mx, __shfl_xor(mx, 8, 32));

      const float mnew = fmaxf(mrow[g], mx);
      const float corr = __expf(mrow[g] - mnew);
      mrow[g] = mnew;

      float rs = 0.f;
#pragma unroll
      for (int j = 0; j < 4; ++j) {
        const float p = __expf(sacc[j][g] - mnew);
        rs += p;
        Pb[(wave * 16 + g + (half << 3)) * LDV + j * 16 + r] = (bf16)p;
      }
      rs += __shfl_xor(rs, 1, 32);
      rs += __shfl_xor(rs, 2, 32);
      rs += __shfl_xor(rs, 4, 32);
      rs += __shfl_xor(rs, 8, 32);
      lrow[g] = lrow[g] * corr + rs;
#pragma unroll
      for (int t = 0; t < 8; ++t) Oacc[t][g] *= corr;
    }
    __syncthreads();  // Pb visible / dscnt settled before re-reading

    // --- O += P * V : 8 HD tiles x 2 K-steps = 16 WMMAs ---
    const v16bf aP0 = load_frag(&Pb[(wave * 16 + r) * LDV + 0], half);
    const v16bf aP1 = load_frag(&Pb[(wave * 16 + r) * LDV + 32], half);
#pragma unroll
    for (int t = 0; t < 8; ++t) {
      Oacc[t] = wmma_bf16(aP0, load_frag(&VT[(t * 16 + r) * LDV + 0], half), Oacc[t]);
      Oacc[t] = wmma_bf16(aP1, load_frag(&VT[(t * 16 + r) * LDV + 32], half), Oacc[t]);
    }
    __syncthreads();  // before next K/V staging overwrites LDS
  }

  // --- normalize and store attn output (bf16, [B,S,NH*HD] layout) ---
#pragma unroll
  for (int g = 0; g < 8; ++g) {
    const float inv = 1.0f / lrow[g];
    const size_t row = (size_t)(bb * Sc + qt * 64 + wave * 16 + g + (half << 3)) * Dc + h * HDc;
#pragma unroll
    for (int t = 0; t < 8; ++t) Og[row + t * 16 + r] = (bf16)(Oacc[t][g] * inv);
  }
}

// ---------------------------------------------------------------------------
extern "C" void kernel_launch(void* const* d_in, const int* in_sizes, int n_in,
                              void* d_out, int out_size, void* d_ws, size_t ws_size,
                              hipStream_t stream) {
  (void)in_sizes; (void)n_in; (void)out_size; (void)ws_size;
  const float* x  = (const float*)d_in[0];
  const float* Wq = (const float*)d_in[1];
  const float* Wk = (const float*)d_in[2];
  const float* Wv = (const float*)d_in[3];
  const float* Wo = (const float*)d_in[4];
  float* out = (float*)d_out;

  // Workspace: Q,K,V,attn in bf16 -> 4 * 16.78M elems * 2B = 128 MiB
  const size_t per = (size_t)Bc * Sc * Dc;
  bf16* Qb = (bf16*)d_ws;
  bf16* Kb = Qb + per;
  bf16* Vb = Kb + per;
  bf16* Ab = Vb + per;

  const int M = Bc * Sc;                     // 8192
  dim3 ggrid(Dc / 128, M / 128);             // (16, 64)

  wmma_gemm_kernel<false, true><<<ggrid, 256, 0, stream>>>(x, Wq, Qb, M, Dc, Dc);
  wmma_gemm_kernel<false, true><<<ggrid, 256, 0, stream>>>(x, Wk, Kb, M, Dc, Dc);
  wmma_gemm_kernel<false, true><<<ggrid, 256, 0, stream>>>(x, Wv, Vb, M, Dc, Dc);

  flash_attn_kernel<<<dim3(Bc * NHc * (Sc / 64)), 128, 0, stream>>>(Qb, Kb, Vb, Ab);

  wmma_gemm_kernel<true, false><<<ggrid, 256, 0, stream>>>(Ab, Wo, out, M, Dc, Dc);
}
